// SelfAttention_74509092651427
// MI455X (gfx1250) — matrix operand order repaired and checked by reading, not compiled, not run on hardware
//
#include <hip/hip_runtime.h>
#include <hip/hip_bf16.h>

// Problem constants (match reference)
#define BB 8
#define CC 512
#define WW 2048
#define CR 64   // C / R

typedef __attribute__((ext_vector_type(16))) __bf16       bf16x16;
typedef __attribute__((ext_vector_type(8)))  float        f32x8;
typedef __attribute__((ext_vector_type(4)))  float        f32x4;
typedef __attribute__((ext_vector_type(4)))  unsigned int u32x4;

union FragU { bf16x16 v; u32x4 q[2]; };
union Pk8U  { u32x4 q; __bf16 h[8]; };
union Pk16U { u32x4 q[2]; __bf16 h[16]; };

// ---------------------------------------------------------------------------
// WMMA fragment loaders (CDNA5 wave32 layouts, ISA 7.12.2)
// A 16x32 bf16: lane m=lane&15; lanes0-15 K-halves {0..7,16..23},
//               lanes16-31 {8..15,24..31} -> two contiguous 16B runs per lane.
// B 32x16 bf16 (n-major source): lane n=lane&15; lanes0-15 K {0..7,16..23},
//               lanes16-31 {8..15,24..31} -> two contiguous 16B runs per lane.
// ---------------------------------------------------------------------------
__device__ inline void load_a_bf16_vec(bf16x16& a, const __bf16* src, int lda, int lane) {
  int m  = lane & 15;
  int kh = (lane >> 4) * 8;
  const __bf16* row = src + (size_t)m * lda + kh;
  FragU u;
  u.q[0] = *(const u32x4*)(row);        // K = kh .. kh+7
  u.q[1] = *(const u32x4*)(row + 16);   // K = 16+kh .. 16+kh+7
  a = u.v;
}

__device__ inline void load_b_bf16_nmajor_vec(bf16x16& b, const __bf16* src, int ldn, int lane) {
  int n  = lane & 15;
  int kb = (lane >> 4) * 8;
  const __bf16* col = src + (size_t)n * ldn + kb;
  FragU u;
  u.q[0] = *(const u32x4*)(col);
  u.q[1] = *(const u32x4*)(col + 16);
  b = u.v;
}

// A 16x32 from f32 row-major source (vectorized 16B loads + cvt to bf16)
__device__ inline void load_a_f32_vec(bf16x16& a, const float* src, int lda, int lane) {
  int m  = lane & 15;
  int kh = (lane >> 4) * 8;
  const float* row = src + (size_t)m * lda + kh;
  f32x4 f0 = *(const f32x4*)(row);
  f32x4 f1 = *(const f32x4*)(row + 4);
  f32x4 f2 = *(const f32x4*)(row + 16);
  f32x4 f3 = *(const f32x4*)(row + 20);
#pragma unroll
  for (int e = 0; e < 4; ++e) {
    a[e]      = (__bf16)f0[e];
    a[4 + e]  = (__bf16)f1[e];
    a[8 + e]  = (__bf16)f2[e];
    a[12 + e] = (__bf16)f3[e];
  }
}

// B 32x16 from f32 K-major source (element (k,n) at src[k*ld+n]; 16 lane-coalesced loads)
__device__ inline void load_b_f32_kmajor(bf16x16& b, const float* src, int ld, int lane) {
  int n  = lane & 15;
  int kb = (lane >> 4) * 8;
#pragma unroll
  for (int i = 0; i < 8; ++i) {
    b[2 * i]     = (__bf16)src[(size_t)(kb + 2 * i + (i >= 4 ? 8 : 0)) * ld + n];
    b[2 * i + 1] = (__bf16)src[(size_t)(kb + 2 * i + 1 + (i >= 4 ? 8 : 0)) * ld + n];
  }
}

__device__ inline f32x8 wmma_bf16(const bf16x16& a, const bf16x16& b, const f32x8& c) {
  return __builtin_amdgcn_wmma_f32_16x16x32_bf16(false, a, false, b, (short)0, c, false, false);
}

// ---------------------------------------------------------------------------
// Kernel 1: QKV projections. One 32(o)x16(w) block per wave (2 WMMA tiles
// share each B fragment of x). q stored transposed & pre-scaled, k transposed,
// v stored [C,W].
// ---------------------------------------------------------------------------
__global__ void qkv_kernel(const float* __restrict__ x,
                           const float* __restrict__ wq, const float* __restrict__ bq,
                           const float* __restrict__ wk, const float* __restrict__ bk,
                           const float* __restrict__ wv, const float* __restrict__ bv,
                           __bf16* __restrict__ qT, __bf16* __restrict__ kT,
                           __bf16* __restrict__ v) {
  const int wave = blockIdx.x * (blockDim.x >> 5) + (threadIdx.x >> 5);
  const int lane = threadIdx.x & 31;
  const int WT = WW / 16;                      // 128 w-tiles
  const int perB = (2 + 2 + 16) * WT;          // q(2 o-blocks)+k(2)+v(16) = 2560

  int b = wave / perB;
  int t = wave % perB;
  const float* wgt;
  const float* bias;
  int kind;
  if (t < 2 * WT)        { kind = 0; wgt = wq; bias = bq; }
  else if (t < 4 * WT)   { kind = 1; wgt = wk; bias = bk; t -= 2 * WT; }
  else                   { kind = 2; wgt = wv; bias = bv; t -= 4 * WT; }
  const int oBase = (t / WT) * 32;
  const int wBase = (t % WT) * 16;
  const float* xb = x + (size_t)b * CC * WW;

  f32x8 acc0 = {}, acc1 = {};
  for (int c = 0; c < CC; c += 32) {
    bf16x16 a0, a1, bf;
    load_b_f32_kmajor(bf, xb + (size_t)c * WW + wBase, WW, lane);
    load_a_f32_vec(a0, wgt + (size_t)oBase * CC + c, CC, lane);
    load_a_f32_vec(a1, wgt + (size_t)(oBase + 16) * CC + c, CC, lane);
    acc0 = wmma_bf16(a0, bf, acc0);
    acc1 = wmma_bf16(a1, bf, acc1);
  }

  const int n = lane & 15, hi = lane >> 4;
#pragma unroll
  for (int sub = 0; sub < 2; ++sub) {
    const f32x8& acc = sub ? acc1 : acc0;
    int oT = oBase + sub * 16;
#pragma unroll
    for (int r = 0; r < 8; ++r) {
      int m = hi * 8 + r;
      float val = acc[r] + bias[oT + m];
      if (kind == 0)
        qT[((size_t)b * WW + wBase + n) * CR + oT + m] = (__bf16)(val * 0.125f);
      else if (kind == 1)
        kT[((size_t)b * WW + wBase + n) * CR + oT + m] = (__bf16)val;
      else
        v[((size_t)b * CC + oT + m) * WW + wBase + n] = (__bf16)val;
    }
  }
}

// ---------------------------------------------------------------------------
// Kernel 2: scores[b,i,j] = sum_d qT[b,i,d]*kT[b,j,d].  32x32 block per wave
// (2x2 tiles, fragments reused twice).
// ---------------------------------------------------------------------------
__global__ void scores_kernel(const __bf16* __restrict__ qT,
                              const __bf16* __restrict__ kT,
                              __bf16* __restrict__ scores) {
  const int wave = blockIdx.x * (blockDim.x >> 5) + (threadIdx.x >> 5);
  const int lane = threadIdx.x & 31;
  const int JB = WW / 32;           // 64
  const int perB = JB * JB;         // 4096
  int b = wave / perB;
  int t = wave % perB;
  const int iBase = (t / JB) * 32;
  const int jBase = (t % JB) * 32;
  const __bf16* qb = qT + (size_t)b * WW * CR;
  const __bf16* kb = kT + (size_t)b * WW * CR;

  f32x8 acc00 = {}, acc01 = {}, acc10 = {}, acc11 = {};
#pragma unroll
  for (int d = 0; d < CR; d += 32) {
    bf16x16 a0, a1, b0, b1;
    load_a_bf16_vec(a0, qb + (size_t)iBase * CR + d, CR, lane);
    load_a_bf16_vec(a1, qb + (size_t)(iBase + 16) * CR + d, CR, lane);
    load_b_bf16_nmajor_vec(b0, kb + (size_t)jBase * CR + d, CR, lane);
    load_b_bf16_nmajor_vec(b1, kb + (size_t)(jBase + 16) * CR + d, CR, lane);
    acc00 = wmma_bf16(a0, b0, acc00);
    acc01 = wmma_bf16(a0, b1, acc01);
    acc10 = wmma_bf16(a1, b0, acc10);
    acc11 = wmma_bf16(a1, b1, acc11);
  }

  const int n = lane & 15, hi = lane >> 4;
#pragma unroll
  for (int ti = 0; ti < 2; ++ti) {
#pragma unroll
    for (int tj = 0; tj < 2; ++tj) {
      const f32x8& acc = ti ? (tj ? acc11 : acc10) : (tj ? acc01 : acc00);
#pragma unroll
      for (int r = 0; r < 8; ++r) {
        int m = hi * 8 + r;
        scores[((size_t)b * WW + iBase + ti * 16 + m) * WW + jBase + tj * 16 + n] = (__bf16)acc[r];
      }
    }
  }
}

// ---------------------------------------------------------------------------
// Kernel 3: row softmax + transpose via LDS.
// Block handles 16 rows i (16 threads per row). Global scores read once into
// a 16x2048 transposed LDS tile, exp in LDS, then coalesced 32B stores of
// attnT[b][j][i0..i0+15].  Dynamic LDS: 64KB tile + reduction scratch.
// ---------------------------------------------------------------------------
__global__ void softmax_kernel(const __bf16* __restrict__ scores,
                               __bf16* __restrict__ attnT) {
  extern __shared__ char smem[];
  __bf16* tile  = (__bf16*)smem;                           // [WW][16]
  float*  red   = (float*)(smem + (size_t)WW * 16 * 2);    // [256]
  float*  inv_s = red + 256;                               // [16]

  const int blocksPerB = WW / 16;   // 128
  const int b   = blockIdx.x / blocksPerB;
  const int i0  = (blockIdx.x % blocksPerB) * 16;
  const int tid = threadIdx.x;
  const int r   = tid >> 4;         // row in block 0..15
  const int c   = tid & 15;         // j-chunk 0..15 (128 j each)

  const __bf16* row = scores + ((size_t)b * WW + i0 + r) * WW + c * 128;
  const u32x4* rp = (const u32x4*)row;

  // pass 1: global -> transposed LDS tile, tracking max
  float lmax = -3.0e30f;
#pragma unroll
  for (int t4 = 0; t4 < 16; ++t4) {
    Pk8U u;
    u.q = rp[t4];
    int jb = c * 128 + t4 * 8;
#pragma unroll
    for (int e = 0; e < 8; ++e) {
      lmax = fmaxf(lmax, (float)u.h[e]);
      tile[(size_t)(jb + e) * 16 + r] = u.h[e];
    }
  }
  red[tid] = lmax;
  __syncthreads();
  float rmax = red[r * 16];
#pragma unroll
  for (int e = 1; e < 16; ++e) rmax = fmaxf(rmax, red[r * 16 + e]);
  __syncthreads();

  // pass 2: exp in place + local sum
  float lsum = 0.0f;
#pragma unroll
  for (int t4 = 0; t4 < 16; ++t4) {
    int jb = c * 128 + t4 * 8;
#pragma unroll
    for (int e = 0; e < 8; ++e) {
      size_t idx = (size_t)(jb + e) * 16 + r;
      float f = __expf((float)tile[idx] - rmax);
      lsum += f;
      tile[idx] = (__bf16)f;
    }
  }
  red[tid] = lsum;
  __syncthreads();
  float rsum = 0.0f;
#pragma unroll
  for (int e = 0; e < 16; ++e) rsum += red[r * 16 + e];
  if (c == 0) inv_s[r] = 1.0f / rsum;
  __syncthreads();

  // pass 3: normalized LDS tile -> attnT, 32B contiguous per j
  __bf16* dstBase = attnT + (size_t)b * WW * WW + i0;
  for (int j = tid; j < WW; j += 256) {
    const __bf16* tp = &tile[(size_t)j * 16];
    Pk16U o;
#pragma unroll
    for (int rr = 0; rr < 16; ++rr)
      o.h[rr] = (__bf16)((float)tp[rr] * inv_s[rr]);
    u32x4* dst = (u32x4*)(dstBase + (size_t)j * WW);
    dst[0] = o.q[0];
    dst[1] = o.q[1];
  }
}

// ---------------------------------------------------------------------------
// Kernel 4: context[b,c,j] = sum_i v[b,c,i]*attn[b,i,j]; out = context + x.
// 32x32 block per wave (2x2 tiles), K-loop over i unrolled 2x.
// ---------------------------------------------------------------------------
__global__ void context_kernel(const __bf16* __restrict__ v,
                               const __bf16* __restrict__ attnT,
                               const float* __restrict__ x,
                               float* __restrict__ out) {
  const int wave = blockIdx.x * (blockDim.x >> 5) + (threadIdx.x >> 5);
  const int lane = threadIdx.x & 31;
  const int JB = WW / 32;             // 64
  const int perB = (CC / 32) * JB;    // 1024
  int b = wave / perB;
  int t = wave % perB;
  const int cBase = (t / JB) * 32;
  const int jBase = (t % JB) * 32;
  const __bf16* vb = v + (size_t)b * CC * WW;
  const __bf16* ab = attnT + (size_t)b * WW * WW;

  f32x8 acc00 = {}, acc01 = {}, acc10 = {}, acc11 = {};
#pragma unroll 2
  for (int i = 0; i < WW; i += 32) {
    bf16x16 a0, a1, b0, b1;
    load_a_bf16_vec(a0, vb + (size_t)cBase * WW + i, WW, lane);
    load_a_bf16_vec(a1, vb + (size_t)(cBase + 16) * WW + i, WW, lane);
    load_b_bf16_nmajor_vec(b0, ab + (size_t)jBase * WW + i, WW, lane);
    load_b_bf16_nmajor_vec(b1, ab + (size_t)(jBase + 16) * WW + i, WW, lane);
    acc00 = wmma_bf16(a0, b0, acc00);
    acc01 = wmma_bf16(a0, b1, acc01);
    acc10 = wmma_bf16(a1, b0, acc10);
    acc11 = wmma_bf16(a1, b1, acc11);
  }

  const int n = lane & 15, hi = lane >> 4;
#pragma unroll
  for (int ti = 0; ti < 2; ++ti) {
#pragma unroll
    for (int tj = 0; tj < 2; ++tj) {
      const f32x8& acc = ti ? (tj ? acc11 : acc10) : (tj ? acc01 : acc00);
#pragma unroll
      for (int r = 0; r < 8; ++r) {
        int m = hi * 8 + r;
        size_t idx = ((size_t)b * CC + cBase + ti * 16 + m) * WW + jBase + tj * 16 + n;
        out[idx] = acc[r] + x[idx];
      }
    }
  }
}

// ---------------------------------------------------------------------------
// Launcher. Workspace layout (bytes):
//   [0,   2M)  qT  bf16 [B][W][CR]
//   [2M,  4M)  kT  bf16 [B][W][CR]
//   [4M, 20M)  v   bf16 [B][C][W]
//   [20M,84M)  scores bf16 [B][W][W]
//   [84M,148M) attnT  bf16 [B][W][W]
// ---------------------------------------------------------------------------
extern "C" void kernel_launch(void* const* d_in, const int* in_sizes, int n_in,
                              void* d_out, int out_size, void* d_ws, size_t ws_size,
                              hipStream_t stream) {
  const float* x  = (const float*)d_in[0];
  const float* wq = (const float*)d_in[1];
  const float* bq = (const float*)d_in[2];
  const float* wk = (const float*)d_in[3];
  const float* bk = (const float*)d_in[4];
  const float* wv = (const float*)d_in[5];
  const float* bv = (const float*)d_in[6];
  float* out = (float*)d_out;

  char* ws = (char*)d_ws;
  const size_t MB = 1024ull * 1024ull;
  __bf16* qT = (__bf16*)(ws);
  __bf16* kT = (__bf16*)(ws + 2 * MB);
  __bf16* v  = (__bf16*)(ws + 4 * MB);
  __bf16* sc = (__bf16*)(ws + 20 * MB);
  __bf16* at = (__bf16*)(ws + 84 * MB);

  // Kernel 1: 8 * 2560 waves, 8 waves/block
  qkv_kernel<<<2560, 256, 0, stream>>>(x, wq, bq, wk, bk, wv, bv, qT, kT, v);
  // Kernel 2: 8 * 4096 waves
  scores_kernel<<<4096, 256, 0, stream>>>(qT, kT, sc);
  // Kernel 3: one block per 16 rows; 64KB tile + 1KB reduction scratch
  const size_t smax = (size_t)WW * 16 * 2 + 256 * 4 + 16 * 4;
  softmax_kernel<<<BB * (WW / 16), 256, smax, stream>>>(sc, at);
  // Kernel 4: 8 * 1024 waves
  context_kernel<<<1024, 256, 0, stream>>>(v, at, x, out);
}
